// CRFDecoder_78503412236940
// MI455X (gfx1250) — matrix operand order repaired and checked by reading, not compiled, not run on hardware
//
#include <hip/hip_runtime.h>
#include <cstdint>
#include <cstddef>

// ---------------------------------------------------------------------------
// CRF decoder for MI455X (gfx1250, wave32, WMMA)
//   B=32, T=512, H=2048, K=128
// ---------------------------------------------------------------------------

#define HIDDEN 2048
#define NTAG   128
#define SEQ    512
#define BATCH  32
#define NEG_INF (-__builtin_inff())

typedef __attribute__((ext_vector_type(16))) __bf16 v16bf;
typedef __attribute__((ext_vector_type(8)))  float  v8f;

// ---------------------------------------------------------------------------
// Kernel 1: split W (fp32, [H,K] row-major) into transposed bf16 hi/lo planes
//           wt_hi/wt_lo are [K][H] so WMMA B-fragments load contiguously.
// ---------------------------------------------------------------------------
__global__ __launch_bounds__(256) void wsplit_kernel(
    const float* __restrict__ W, __bf16* __restrict__ wt_hi,
    __bf16* __restrict__ wt_lo) {
  int idx = blockIdx.x * 256 + threadIdx.x;       // 2048*128 = 262144 elements
  int h = idx >> 7;
  int k = idx & 127;
  float w  = W[idx];
  __bf16 hi = (__bf16)w;
  float  lo = w - (float)hi;
  wt_hi[(size_t)k * HIDDEN + h] = hi;
  wt_lo[(size_t)k * HIDDEN + h] = (__bf16)lo;
}

// ---------------------------------------------------------------------------
// Kernel 2: logits = hiddens @ W + b via bf16x3 split WMMA.
//   grid = 128 blocks (128 M-rows each), 256 threads = 8 waves.
//   wave w -> rows [blk*128 + w*16, +16), all 128 columns (8 N-tiles).
// ---------------------------------------------------------------------------
__global__ __launch_bounds__(256) void logits_gemm_kernel(
    const float* __restrict__ hid, const __bf16* __restrict__ wt_hi,
    const __bf16* __restrict__ wt_lo, const float* __restrict__ bias,
    float* __restrict__ logits) {
  const int wave    = threadIdx.x >> 5;
  const int lane    = threadIdx.x & 31;
  const int half    = lane >> 4;   // K-half selector for A/B fragments
  const int r       = lane & 15;   // M (for A) / N (for B,C) within tile
  const int rowBase = blockIdx.x * 128 + wave * 16;

  v8f acc[8] = {};

  // A-fragment base: element i -> K = kc + half*16 + i
  const float* aRow = hid + (size_t)(rowBase + r) * HIDDEN + half * 16;

  for (int kc = 0; kc < HIDDEN; kc += 32) {
    // prefetch next A chunk (emits global_prefetch_b8)
    __builtin_prefetch((const void*)(aRow + kc + 32), 0, 1);

    v8f f0 = *(const v8f*)(aRow + kc);
    v8f f1 = *(const v8f*)(aRow + kc + 8);
    v16bf a_hi, a_lo;
#pragma unroll
    for (int e = 0; e < 8; ++e) {
      float f   = f0[e];
      __bf16 h0 = (__bf16)f;
      a_hi[e]     = h0;
      a_lo[e]     = (__bf16)(f - (float)h0);
      float g   = f1[e];
      __bf16 h1 = (__bf16)g;
      a_hi[e + 8] = h1;
      a_lo[e + 8] = (__bf16)(g - (float)h1);
    }

#pragma unroll
    for (int nt = 0; nt < 8; ++nt) {
      // B-fragment: lane holds column nt*16+r, 16 consecutive K at half*16
      const __bf16* bh = wt_hi + (size_t)(nt * 16 + r) * HIDDEN + kc + half * 16;
      const __bf16* bl = wt_lo + (size_t)(nt * 16 + r) * HIDDEN + kc + half * 16;
      v16bf b_hi = *(const v16bf*)bh;
      v16bf b_lo = *(const v16bf*)bl;
      // bf16x3: hi*hi + lo*hi + hi*lo  (~fp32 accuracy)
      acc[nt] = __builtin_amdgcn_wmma_f32_16x16x32_bf16(
          false, a_hi, false, b_hi, (short)0, acc[nt], false, false);
      acc[nt] = __builtin_amdgcn_wmma_f32_16x16x32_bf16(
          false, a_lo, false, b_hi, (short)0, acc[nt], false, false);
      acc[nt] = __builtin_amdgcn_wmma_f32_16x16x32_bf16(
          false, a_hi, false, b_lo, (short)0, acc[nt], false, false);
    }
  }

  // C layout: VGPR v -> M = v + 8*half, N = r
#pragma unroll
  for (int nt = 0; nt < 8; ++nt) {
    int col  = nt * 16 + r;
    float bv = bias[col];
#pragma unroll
    for (int v = 0; v < 8; ++v) {
      int row = rowBase + v + 8 * half;
      logits[(size_t)row * NTAG + col] = acc[nt][v] + bv;
    }
  }
}

// ---------------------------------------------------------------------------
// Kernel 3: fused Viterbi max/argmax + forward logsumexp recurrences.
//   grid = 32 blocks (one batch each), 512 threads: j = tid&127 (output tag),
//   s = tid>>7 (one of 4 slices of the 128-wide i-reduction).
//   transitions (64 KB) staged in dynamic LDS (WGP has 320 KB); two-pass LSE
//   with 4-way split accumulator chains to kill trans-unit latency chains.
// ---------------------------------------------------------------------------
__global__ __launch_bounds__(512) void crf_recurrence_kernel(
    const float* __restrict__ logits, const float* __restrict__ trans,
    const int* __restrict__ mask, const float* __restrict__ start_t,
    const float* __restrict__ end_t, unsigned char* __restrict__ bp,
    float* __restrict__ logZ, int* __restrict__ last_tag) {
  const int b   = blockIdx.x;
  const int tid = threadIdx.x;
  const int j   = tid & 127;
  const int s   = tid >> 7;  // slice 0..3, covers i in [s*32, s*32+32)

  extern __shared__ float tr_lds[];        // NTAG*NTAG fp32 = 64 KB (dynamic)
  __shared__ float s_score[NTAG];
  __shared__ float s_alpha[NTAG];
  __shared__ float r_best[4][NTAG];
  __shared__ int   r_arg[4][NTAG];
  __shared__ float r_m[4][NTAG];
  __shared__ float r_s[4][NTAG];

  const float* lg = logits + (size_t)b * SEQ * NTAG;

  // stage the full transition matrix into LDS once
  for (int idx = tid; idx < NTAG * NTAG; idx += 512) tr_lds[idx] = trans[idx];

  if (s == 0) {
    float v    = start_t[j] + lg[j];
    s_score[j] = v;
    s_alpha[j] = v;
  }
  __syncthreads();

  const int ibase = s * 32;

  for (int t = 1; t < SEQ; ++t) {
    // ---- pass 1: Viterbi max/argmax + LSE max, 4 independent chains ----
    float b0 = NEG_INF, b1 = NEG_INF, b2 = NEG_INF, b3 = NEG_INF;
    int   x0 = 0, x1 = 0, x2 = 0, x3 = 0;
    float m0 = NEG_INF, m1 = NEG_INF, m2 = NEG_INF, m3 = NEG_INF;
#pragma unroll
    for (int ii = 0; ii < 32; ii += 4) {
      int i = ibase + ii;
      float t0 = tr_lds[(i + 0) * NTAG + j];
      float t1 = tr_lds[(i + 1) * NTAG + j];
      float t2 = tr_lds[(i + 2) * NTAG + j];
      float t3 = tr_lds[(i + 3) * NTAG + j];
      float sv0 = s_score[i + 0] + t0;
      float sv1 = s_score[i + 1] + t1;
      float sv2 = s_score[i + 2] + t2;
      float sv3 = s_score[i + 3] + t3;
      if (sv0 > b0) { b0 = sv0; x0 = i + 0; }
      if (sv1 > b1) { b1 = sv1; x1 = i + 1; }
      if (sv2 > b2) { b2 = sv2; x2 = i + 2; }
      if (sv3 > b3) { b3 = sv3; x3 = i + 3; }
      m0 = fmaxf(m0, s_alpha[i + 0] + t0);
      m1 = fmaxf(m1, s_alpha[i + 1] + t1);
      m2 = fmaxf(m2, s_alpha[i + 2] + t2);
      m3 = fmaxf(m3, s_alpha[i + 3] + t3);
    }
    // merge chains (index-aware: first-max semantics like jnp.argmax)
    float best = b0; int bi = x0;
    if (b1 > best || (b1 == best && x1 < bi)) { best = b1; bi = x1; }
    if (b2 > best || (b2 == best && x2 < bi)) { best = b2; bi = x2; }
    if (b3 > best || (b3 == best && x3 < bi)) { best = b3; bi = x3; }
    float m = fmaxf(fmaxf(m0, m1), fmaxf(m2, m3));

    // ---- pass 2: sum exp(alpha+tr-m), 1 exp per i, 4 partial sums ----
    float p0 = 0.f, p1 = 0.f, p2 = 0.f, p3 = 0.f;
#pragma unroll
    for (int ii = 0; ii < 32; ii += 4) {
      int i = ibase + ii;
      p0 += __expf(s_alpha[i + 0] + tr_lds[(i + 0) * NTAG + j] - m);
      p1 += __expf(s_alpha[i + 1] + tr_lds[(i + 1) * NTAG + j] - m);
      p2 += __expf(s_alpha[i + 2] + tr_lds[(i + 2) * NTAG + j] - m);
      p3 += __expf(s_alpha[i + 3] + tr_lds[(i + 3) * NTAG + j] - m);
    }
    float ssum = (p0 + p1) + (p2 + p3);

    r_best[s][j] = best;
    r_arg[s][j]  = bi;
    r_m[s][j]    = m;
    r_s[s][j]    = ssum;
    __syncthreads();

    if (s == 0) {
      float bb = r_best[0][j];
      int   aa = r_arg[0][j];
#pragma unroll
      for (int k = 1; k < 4; ++k) {   // slices are in i-order: strict > keeps first
        if (r_best[k][j] > bb) { bb = r_best[k][j]; aa = r_arg[k][j]; }
      }
      float mm = r_m[0][j];
      float sS = r_s[0][j];
#pragma unroll
      for (int k = 1; k < 4; ++k) {
        float mk = r_m[k][j];
        float nm = fmaxf(mm, mk);
        sS = sS * __expf(mm - nm) + r_s[k][j] * __expf(mk - nm);
        mm = nm;
      }
      float e  = lg[(size_t)t * NTAG + j];
      int   on = mask[b * SEQ + t];
      float ns = on ? (bb + e) : s_score[j];
      float na = on ? (mm + __logf(sS) + e) : s_alpha[j];
      unsigned char nb = on ? (unsigned char)aa : (unsigned char)j;
      bp[((size_t)b * SEQ + t) * NTAG + j] = nb;
      s_score[j] = ns;
      s_alpha[j] = na;
    }
    __syncthreads();
  }

  if (tid == 0) {
    // logZ = logsumexp_j(alpha[j] + end_t[j])
    float m = NEG_INF, ss = 0.0f;
    for (int jj = 0; jj < NTAG; ++jj) {
      float v  = s_alpha[jj] + end_t[jj];
      float nm = fmaxf(m, v);
      ss = ss * __expf(m - nm) + __expf(v - nm);
      m  = nm;
    }
    logZ[b] = m + __logf(ss);
    // last_tag = argmax_j(score[j] + end_t[j])  (first max, like jnp.argmax)
    float bb = NEG_INF;
    int   aa = 0;
    for (int jj = 0; jj < NTAG; ++jj) {
      float v = s_score[jj] + end_t[jj];
      if (v > bb) { bb = v; aa = jj; }
    }
    last_tag[b] = aa;
  }
}

// ---------------------------------------------------------------------------
// Kernel 4: gold-path score, backtrace, loss. One block of 32 threads (1/batch).
//   d_out layout: decode (B*T floats, tag values) then loss at index B*T.
// ---------------------------------------------------------------------------
__global__ __launch_bounds__(32) void crf_finalize_kernel(
    const float* __restrict__ logits, const float* __restrict__ trans,
    const int* __restrict__ mask, const int* __restrict__ labels,
    const float* __restrict__ start_t, const float* __restrict__ end_t,
    const float* __restrict__ logZ, const int* __restrict__ last_tag,
    const unsigned char* __restrict__ bp, float* __restrict__ outf) {
  const int b = threadIdx.x;
  __shared__ float sll[BATCH];

  const float* lg = logits + (size_t)b * SEQ * NTAG;

  // ---- numerator score ----
  int   tag0 = labels[b * SEQ];
  float sc   = start_t[tag0] + lg[tag0];
  int   prev = tag0;
  int   msum = mask[b * SEQ];
  for (int t = 1; t < SEQ; ++t) {
    int   cur = labels[b * SEQ + t];
    float mf  = (float)mask[b * SEQ + t];
    sc += (trans[prev * NTAG + cur] + lg[(size_t)t * NTAG + cur]) * mf;
    msum += mask[b * SEQ + t];
    prev = cur;
  }
  int last_idx = msum - 1;
  sc += end_t[labels[b * SEQ + last_idx]];
  sll[b] = sc - logZ[b];

  // ---- Viterbi backtrace ----
  int tg = last_tag[b];
  outf[b * SEQ + (SEQ - 1)] = (float)tg;
  for (int t = SEQ - 1; t >= 1; --t) {
    tg = (int)bp[((size_t)b * SEQ + t) * NTAG + tg];
    outf[b * SEQ + (t - 1)] = (float)tg;
  }

  __syncthreads();
  if (b == 0) {
    float sum = 0.0f;
    for (int i = 0; i < BATCH; ++i) sum += sll[i];
    outf[BATCH * SEQ] = -(sum / (float)BATCH);  // loss = -mean(llh)
  }
}

// ---------------------------------------------------------------------------
// Launcher
// ---------------------------------------------------------------------------
extern "C" void kernel_launch(void* const* d_in, const int* in_sizes, int n_in,
                              void* d_out, int out_size, void* d_ws,
                              size_t ws_size, hipStream_t stream) {
  const float* hiddens = (const float*)d_in[0];
  const int*   mask    = (const int*)d_in[1];
  const int*   labels  = (const int*)d_in[2];
  const float* W       = (const float*)d_in[3];
  const float* bias    = (const float*)d_in[4];
  const float* start_t = (const float*)d_in[5];
  const float* end_t   = (const float*)d_in[6];
  const float* trans   = (const float*)d_in[7];

  char* ws = (char*)d_ws;
  // workspace layout (≈11 MB total)
  __bf16* wt_hi  = (__bf16*)(ws + 0);                           // 512 KB
  __bf16* wt_lo  = (__bf16*)(ws + (size_t)512 * 1024);          // 512 KB
  float*  logits = (float*)(ws + (size_t)1024 * 1024);          // 8 MB
  unsigned char* bp = (unsigned char*)(ws + (size_t)9 * 1024 * 1024);  // 2 MB
  float* logZ    = (float*)(ws + (size_t)11 * 1024 * 1024);     // 128 B
  int*   lastTag = (int*)(ws + (size_t)11 * 1024 * 1024 + 256); // 128 B

  float* outf = (float*)d_out;

  hipLaunchKernelGGL(wsplit_kernel, dim3((HIDDEN * NTAG) / 256), dim3(256), 0,
                     stream, W, wt_hi, wt_lo);
  hipLaunchKernelGGL(logits_gemm_kernel, dim3((BATCH * SEQ) / 128), dim3(256),
                     0, stream, hiddens, wt_hi, wt_lo, bias, logits);
  hipLaunchKernelGGL(crf_recurrence_kernel, dim3(BATCH), dim3(512),
                     NTAG * NTAG * sizeof(float), stream, logits, trans, mask,
                     start_t, end_t, bp, logZ, lastTag);
  hipLaunchKernelGGL(crf_finalize_kernel, dim3(1), dim3(BATCH), 0, stream,
                     logits, trans, mask, labels, start_t, end_t, logZ, lastTag,
                     bp, outf);
}
